// AdditiveAttention_24739011625275
// MI455X (gfx1250) — compile-verified, matrix operand order checked by reading, not run on hardware
//
#include <hip/hip_runtime.h>
#include <hip/hip_bf16.h>

typedef __attribute__((ext_vector_type(16))) __bf16 v16bf;
typedef __attribute__((ext_vector_type(8)))  float  v8f;

#define Bsz 8
#define Qsz 256
#define Ksz 256
#define Dsz 128

union bfrag_u {
  v16bf v;
  unsigned int u[8];
};

// pack two f32 as bf16 pair into one dword (lowers to v_cvt_pk_bf16_f32)
__device__ __forceinline__ unsigned int pack_bf16(float lo, float hi) {
  union { __bf16 h[2]; unsigned int u; } un;
  un.h[0] = (__bf16)lo;
  un.h[1] = (__bf16)hi;
  return un.u;
}

// fast tanh: tanh(x) = 1 - 2/(exp2(2*log2e*x)+1); correct limits at +/-inf
__device__ __forceinline__ float tanh_fast(float x) {
  float t = __builtin_amdgcn_exp2f(x * 2.8853900817779268f); // 2*log2(e)
  return fmaf(-2.0f, __builtin_amdgcn_rcpf(t + 1.0f), 1.0f);
}

// K-pair base for 16-bit A/B fragments (ISA 7.12.2, 16-bit A 16x32 layout)
__device__ __forceinline__ int frag_k0(int v, int half) {
  return ((v & 3) << 1) + ((v >> 2) << 4) + (half << 3);
}

// Y(M x 128) = X(M x 128) * W^T, W stored (128 x 128) row-major as (n,k).
// B tile (W rows nTile..nTile+15, all 128 k) staged ONCE per block into LDS,
// pre-packed as bf16 pairs in WMMA B-fragment dword order: wtile[k/2][n].
// 8 waves per block each own one 16x16 C tile (distinct mTile, shared nTile).
__global__ void proj_kernel(const float* __restrict__ X, const float* __restrict__ W,
                            float* __restrict__ Y) {
  __shared__ unsigned int wtile[(Dsz / 2) * 16]; // 64x16 dwords = 4 KB
  const int tid   = threadIdx.x;
  const int lane  = tid & 31;
  const int wave  = tid >> 5;
  const int mTile = (blockIdx.x * 8 + wave) << 4;
  const int nTile = blockIdx.y << 4;
  const int mn    = lane & 15;
  const int half  = lane >> 4;

  for (int i = tid; i < (Dsz / 2) * 16; i += 256) {
    const int kp = i >> 4, n = i & 15;
    const float* wp = W + (size_t)(nTile + n) * Dsz + (kp << 1);
    wtile[i] = pack_bf16(wp[0], wp[1]);
  }
  __syncthreads();

  v8f c = {};
  for (int kb = 0; kb < Dsz; kb += 32) {
    v16bf a;
    bfrag_u bb;
#pragma unroll
    for (int v = 0; v < 8; ++v) {
      const int k0 = frag_k0(v, half);
      const float* ap = X + (size_t)(mTile + mn) * Dsz + kb + k0;
      a[2 * v]     = (__bf16)ap[0];
      a[2 * v + 1] = (__bf16)ap[1];
      bb.u[v] = wtile[(((kb + k0) >> 1) << 4) + mn];
    }
    c = __builtin_amdgcn_wmma_f32_16x16x32_bf16(false, a, false, bb.v, (short)0, c,
                                                false, false);
  }
#pragma unroll
  for (int r = 0; r < 8; ++r)
    Y[(size_t)(mTile + r + (half << 3)) * Dsz + nTile + mn] = c[r];
}

// scores[b,q,k] = sum_h wv[h] * tanh(qp[b,q,h] + kp[b,k,h])
// One 256-thread block per (b, 16q x 16k) tile; tiles staged in LDS,
// rows padded to 132 floats -> the 16 distinct k-rows land in distinct banks.
__global__ void scores_kernel(const float* __restrict__ qp, const float* __restrict__ kp,
                              const float* __restrict__ wv, float* __restrict__ scores) {
  __shared__ __align__(16) float qs[16 * 132];
  __shared__ __align__(16) float ks[16 * 132];
  __shared__ float wvs[Dsz];
  const int tid   = threadIdx.x;
  const int b     = blockIdx.z;
  const int qTile = blockIdx.y << 4;
  const int kTile = blockIdx.x << 4;
  const float* qsrc = qp + ((size_t)b * Qsz + qTile) * Dsz;
  const float* ksrc = kp + ((size_t)b * Ksz + kTile) * Dsz;
  for (int i = tid; i < 512; i += 256) {
    const int row = i >> 5, c4 = (i & 31) << 2;
    float4 vq = *(const float4*)(qsrc + row * Dsz + c4);
    float4 vk = *(const float4*)(ksrc + row * Dsz + c4);
    *(float4*)(&qs[row * 132 + c4]) = vq;
    *(float4*)(&ks[row * 132 + c4]) = vk;
  }
  if (tid < Dsz) wvs[tid] = wv[tid];
  __syncthreads();
  const int ql = tid >> 4, kl = tid & 15;
  const float* qrow = &qs[ql * 132];
  const float* krow = &ks[kl * 132];
  float s = 0.f;
#pragma unroll 8
  for (int h = 0; h < Dsz; ++h)
    s = fmaf(wvs[h], tanh_fast(qrow[h] + krow[h]), s);
  scores[((size_t)b * Qsz + (qTile + ql)) * Ksz + kTile + kl] = s;
}

// softmax over the QUERY axis (faithful to reference: softmax(scores, axis=1)).
// One thread per (b,k) column; lanes cover consecutive k -> coalesced rows.
__global__ void softmax_q_kernel(const float* __restrict__ scores, float* __restrict__ attn) {
  const int b = blockIdx.y;
  const int k = blockIdx.x * blockDim.x + threadIdx.x;
  const float* S = scores + (size_t)b * Qsz * Ksz;
  float*      A  = attn   + (size_t)b * Qsz * Ksz;
  float m = -1e30f;
  for (int q = 0; q < Qsz; ++q) m = fmaxf(m, S[q * Ksz + k]);
  float sum = 0.f;
  for (int q = 0; q < Qsz; ++q) {
    float e = __builtin_amdgcn_exp2f((S[q * Ksz + k] - m) * 1.4426950408889634f);
    A[q * Ksz + k] = e;
    sum += e;
  }
  const float inv = 1.0f / sum;
  for (int q = 0; q < Qsz; ++q) A[q * Ksz + k] *= inv;
}

// out[b,q,v] = sum_k attn[b,q,k] * values[b,k,v]  -> per-batch 256x128x256 GEMM.
// values^T tile (256 k x 16 v cols) staged once per block into LDS as packed
// bf16 pairs in B-fragment dword order: vtile[k/2][n]. Removes the 8x-redundant
// stride-512B per-wave loads of round 1.
__global__ void av_kernel(const float* __restrict__ attn, const float* __restrict__ V,
                          float* __restrict__ out) {
  __shared__ unsigned int vtile[(Ksz / 2) * 16]; // 128x16 dwords = 8 KB
  const int tid   = threadIdx.x;
  const int lane  = tid & 31;
  const int wave  = tid >> 5;
  const int b     = blockIdx.z;
  const int mTile = (blockIdx.x * 8 + wave) << 4;   // q tile
  const int nTile = blockIdx.y << 4;                // v tile
  const int mn    = lane & 15;
  const int half  = lane >> 4;
  const float* A  = attn + (size_t)b * Qsz * Ksz;
  const float* Vb = V    + (size_t)b * Ksz * Dsz;

  for (int i = tid; i < (Ksz / 2) * 16; i += 256) {
    const int kp = i >> 4, n = i & 15;
    const float* vp = Vb + (size_t)(kp << 1) * Dsz + nTile + n;
    vtile[i] = pack_bf16(vp[0], vp[Dsz]);
  }
  __syncthreads();

  v8f c = {};
  for (int kb = 0; kb < Ksz; kb += 32) {
    v16bf a;
    bfrag_u bb;
#pragma unroll
    for (int v = 0; v < 8; ++v) {
      const int k0 = frag_k0(v, half);
      const float* ap = A + (size_t)(mTile + mn) * Ksz + kb + k0;
      a[2 * v]     = (__bf16)ap[0];
      a[2 * v + 1] = (__bf16)ap[1];
      bb.u[v] = vtile[(((kb + k0) >> 1) << 4) + mn];
    }
    c = __builtin_amdgcn_wmma_f32_16x16x32_bf16(false, a, false, bb.v, (short)0, c,
                                                false, false);
  }
#pragma unroll
  for (int r = 0; r < 8; ++r)
    out[(size_t)b * Qsz * Dsz + (size_t)(mTile + r + (half << 3)) * Dsz + nTile + mn] = c[r];
}

extern "C" void kernel_launch(void* const* d_in, const int* in_sizes, int n_in,
                              void* d_out, int out_size, void* d_ws, size_t ws_size,
                              hipStream_t stream) {
  const float* queries = (const float*)d_in[0];
  const float* keys    = (const float*)d_in[1];
  const float* values  = (const float*)d_in[2];
  const float* W_q     = (const float*)d_in[3];
  const float* W_k     = (const float*)d_in[4];
  const float* w_v     = (const float*)d_in[5];
  float* out = (float*)d_out;

  char* ws = (char*)d_ws;
  const size_t projBytes  = (size_t)Bsz * Qsz * Dsz * sizeof(float); // 1 MB
  const size_t scoreBytes = (size_t)Bsz * Qsz * Ksz * sizeof(float); // 2 MB
  float* qp     = (float*)(ws);
  float* kp     = (float*)(ws + projBytes);
  float* scores = (float*)(ws + 2 * projBytes);
  float* attn   = (float*)(ws + 2 * projBytes + scoreBytes);

  // Projections: M = B*Q = 2048 rows. grid.x=16 blocks * 8 waves * 16 rows = 2048.
  proj_kernel<<<dim3(16, Dsz / 16), 256, 0, stream>>>(queries, W_q, qp);
  proj_kernel<<<dim3(16, Dsz / 16), 256, 0, stream>>>(keys,    W_k, kp);

  // Scores + tanh: (K/16, Q/16, B) tiles.
  scores_kernel<<<dim3(Ksz / 16, Qsz / 16, Bsz), 256, 0, stream>>>(qp, kp, w_v, scores);

  // Softmax over query axis: one thread per (b,k) column.
  softmax_q_kernel<<<dim3(Ksz / 256, Bsz), 256, 0, stream>>>(scores, attn);

  // attn @ values per batch.
  av_kernel<<<dim3(Qsz / 16 / 8, Dsz / 16, Bsz), 256, 0, stream>>>(attn, values, out);
}
